// TemporalAttentionStack_58317065945698
// MI455X (gfx1250) — matrix operand order repaired and checked by reading, not compiled, not run on hardware
//
#include <hip/hip_runtime.h>
#include <hip/hip_bf16.h>
#include <math.h>

typedef _Float16 f16;
typedef __attribute__((ext_vector_type(8)))  _Float16 v8h;
typedef __attribute__((ext_vector_type(16))) _Float16 v16h;
typedef __attribute__((ext_vector_type(8)))  float    v8f;
typedef __attribute__((ext_vector_type(4)))  unsigned v4u;
typedef __attribute__((ext_vector_type(8)))  unsigned v8u;

#define DMODEL 768
#define NHEAD  12
#define DHEAD  64
#define TSEQ   64
#define BBATCH 256
#define NLAYER 4
#define DFF    3072
#define BT     (BBATCH*TSEQ)   /* 16384 */

__device__ __forceinline__ float gelu_exact(float x) {
  return 0.5f * x * (1.0f + erff(x * 0.70710678118654752f));
}

// ------------------------------------------------------------------
// TDM: issue a 2-D tensor_load_to_lds (D# per CDNA5 ISA).
// data_size = 2 bytes. Rows of tile_d0 elements, tile_d1 rows, row
// stride stride0 (elements). Wave-level op (EXEC ignored); TENSORcnt.
// ------------------------------------------------------------------
__device__ __forceinline__ void tdm_load_2d(unsigned lds_off, unsigned long long gaddr,
                                            unsigned tensor_d0, unsigned tensor_d1,
                                            unsigned long long stride0,
                                            unsigned tile_d0, unsigned tile_d1) {
  v4u g0;
  g0.x = 1u;                                              // count=1, user mode
  g0.y = lds_off;                                         // lds_addr [63:32]
  g0.z = (unsigned)(gaddr & 0xFFFFFFFFu);                 // global_addr lo
  g0.w = (unsigned)((gaddr >> 32) & 0x01FFFFFFu)          // global_addr [56:32]
         | (2u << 30);                                    // type=2 ("image")
  v8u g1;
  g1[0] = 1u << 16;                                       // data_size=1 (2B), mask=0
  g1[1] = (tensor_d0 & 0xFFFFu) << 16;                    // tensor_dim0[15:0] @63:48
  g1[2] = ((tensor_d0 >> 16) & 0xFFFFu)                   // tensor_dim0[31:16]
        | ((tensor_d1 & 0xFFFFu) << 16);                  // tensor_dim1[15:0]
  g1[3] = ((tensor_d1 >> 16) & 0xFFFFu)                   // tensor_dim1[31:16]
        | ((tile_d0 & 0xFFFFu) << 16);                    // tile_dim0
  g1[4] = (tile_d1 & 0xFFFFu);                            // tile_dim1, tile_dim2=0
  g1[5] = (unsigned)(stride0 & 0xFFFFFFFFu);              // tensor_dim0_stride lo
  g1[6] = (unsigned)((stride0 >> 32) & 0xFFFFu);          // stride hi, dim1_stride=0
  g1[7] = 0u;
  v4u gz = {0u, 0u, 0u, 0u};
  asm volatile("tensor_load_to_lds %0, %1, %2, %3"
               :: "s"(g0), "s"(g1), "s"(gz), "s"(gz)
               : "memory");
}

// ------------------------------------------------------------------
// f32 -> f16 conversion (weights, activations)
// ------------------------------------------------------------------
__global__ void k_cvt16(const float* __restrict__ src, f16* __restrict__ dst, long n) {
  long i = (long)blockIdx.x * blockDim.x + threadIdx.x;
  long stride = (long)gridDim.x * blockDim.x;
  for (; i < n; i += stride) dst[i] = (f16)src[i];
}

// ------------------------------------------------------------------
// x = frame_embeddings + [sin(ts*f) | cos(ts*f)]
// ------------------------------------------------------------------
__global__ void k_embed(const float* __restrict__ frame, const float* __restrict__ ts,
                        float* __restrict__ x) {
  long i = (long)blockIdx.x * blockDim.x + threadIdx.x;
  long n = (long)BT * DMODEL;
  if (i >= n) return;
  long row = i / DMODEL;
  int d = (int)(i - row * DMODEL);
  int j = (d < 384) ? d : (d - 384);
  float f = __expf(-5.9007135f * (float)j * (1.0f / 384.0f)); // log(365.25)
  float a = ts[row] * f;
  float pe = (d < 384) ? __sinf(a) : __cosf(a);
  x[i] = frame[i] + pe;
}

// conf = (1-clip(cloud))*0.95+0.05 ; logconf = log(max(conf,1e-6))
__global__ void k_conf(const float* __restrict__ cloud, float* __restrict__ conf,
                       float* __restrict__ logconf, float* __restrict__ conf_out) {
  int i = blockIdx.x * blockDim.x + threadIdx.x;
  if (i >= BT) return;
  float c = cloud[i];
  c = fminf(fmaxf(c, 0.f), 1.f);
  float w = (1.f - c) * 0.95f + 0.05f;
  conf[i] = w;
  logconf[i] = __logf(fmaxf(w, 1e-6f));
  conf_out[i] = w;
}

// ------------------------------------------------------------------
// LayerNorm over D=768. One row per block, 256 threads * 3 elems.
// ------------------------------------------------------------------
__global__ void __launch_bounds__(256)
k_layernorm(const float* __restrict__ x, const float* __restrict__ g,
            const float* __restrict__ bta, float* __restrict__ o32,
            f16* __restrict__ o16) {
  __shared__ float red[256];
  __shared__ float mv[2];
  long row = blockIdx.x;
  const float* xr = x + row * DMODEL;
  int tid = threadIdx.x;
  float v0 = xr[tid], v1 = xr[tid + 256], v2 = xr[tid + 512];
  red[tid] = v0 + v1 + v2;
  __syncthreads();
  for (int s = 128; s > 0; s >>= 1) { if (tid < s) red[tid] += red[tid + s]; __syncthreads(); }
  if (tid == 0) mv[0] = red[0] * (1.0f / DMODEL);
  __syncthreads();
  float m = mv[0];
  float d0 = v0 - m, d1 = v1 - m, d2 = v2 - m;
  red[tid] = d0 * d0 + d1 * d1 + d2 * d2;
  __syncthreads();
  for (int s = 128; s > 0; s >>= 1) { if (tid < s) red[tid] += red[tid + s]; __syncthreads(); }
  if (tid == 0) mv[1] = rsqrtf(red[0] * (1.0f / DMODEL) + 1e-5f);
  __syncthreads();
  float rinv = mv[1];
#pragma unroll
  for (int j = 0; j < 3; ++j) {
    int d = tid + j * 256;
    float dv = (j == 0 ? d0 : (j == 1 ? d1 : d2));
    float y = dv * rinv * g[d] + bta[d];
    if (o32) o32[row * DMODEL + d] = y;
    if (o16) o16[row * DMODEL + d] = (f16)y;
  }
}

// ------------------------------------------------------------------
// WMMA GEMM, TDM double-buffered LDS pipeline.
// out[m,n] = act( A[m,:] . Bw[n,:] + bias[n] ) (+resid)
// A: [M,K] f16 row-major. Bw: [N,K] f16 row-major (W used as W^T).
// Block = 256 thr = 8 waves; block tile 64(M) x 128(N); wave 16x64.
// K staged in 64-half chunks via tensor_load_to_lds (TENSORcnt).
// Requires M%64==0, N%128==0, K%64==0 (true at all call sites).
// ------------------------------------------------------------------
__global__ void __launch_bounds__(256)
k_gemm(const f16* __restrict__ A, const f16* __restrict__ Bw,
       const float* __restrict__ bias, const float* __restrict__ resid,
       float* __restrict__ o32, f16* __restrict__ o16,
       int M, int N, int K, int do_gelu) {
  __shared__ f16 ldsA[2][64 * 64];    // [row 0..63][k 0..63]
  __shared__ f16 ldsB[2][128 * 64];   // [col 0..127][k 0..63]
  int wid = threadIdx.x >> 5;
  int lane = threadIdx.x & 31;
  int ln = lane & 15, hi = lane >> 4;
  int m0b = blockIdx.y * 64;
  int n0b = blockIdx.x * 128;
  int ar = (wid & 3) * 16 + ln;            // A row within tile (0..63)
  int bc = (wid >> 2) * 64 + ln;           // first B col within tile
  v8f acc0 = {}, acc1 = {}, acc2 = {}, acc3 = {};
  int nk = K >> 6;                          // # 64-half K chunks

  // prologue: stage 0
  if (wid == 0) {
    tdm_load_2d((unsigned)(unsigned long long)(void*)&ldsA[0][0],
                (unsigned long long)(const void*)(A + (long)m0b * K),
                (unsigned)K, (unsigned)M, (unsigned long long)K, 64u, 64u);
    tdm_load_2d((unsigned)(unsigned long long)(void*)&ldsB[0][0],
                (unsigned long long)(const void*)(Bw + (long)n0b * K),
                (unsigned)K, (unsigned)N, (unsigned long long)K, 64u, 128u);
  }

  for (int kc = 0; kc < nk; ++kc) {
    int cur = kc & 1;
    if (wid == 0) {
      __builtin_amdgcn_s_wait_tensorcnt(0);   // stage `cur` resident
      if (kc + 1 < nk) {                       // prefetch next stage
        int nxt = cur ^ 1;
        long koff = (long)(kc + 1) << 6;
        tdm_load_2d((unsigned)(unsigned long long)(void*)&ldsA[nxt][0],
                    (unsigned long long)(const void*)(A + (long)m0b * K + koff),
                    (unsigned)K, (unsigned)M, (unsigned long long)K, 64u, 64u);
        tdm_load_2d((unsigned)(unsigned long long)(void*)&ldsB[nxt][0],
                    (unsigned long long)(const void*)(Bw + (long)n0b * K + koff),
                    (unsigned)K, (unsigned)N, (unsigned long long)K, 64u, 128u);
      }
    }
    __syncthreads();                          // cur tiles visible to all waves

    const f16* Abuf = &ldsA[cur][0];
    const f16* Bbuf = &ldsB[cur][0];
#pragma unroll
    for (int ks = 0; ks < 2; ++ks) {
      int k_in = ks * 32;
      // A fragment (16x32): lane row ar, 16 contiguous K halves
      v8h a0 = *(const v8h*)(Abuf + ar * 64 + k_in + hi * 16);
      v8h a1 = *(const v8h*)(Abuf + ar * 64 + k_in + hi * 16 + 8);
      v16h av = __builtin_shufflevector(a0, a1, 0, 1, 2, 3, 4, 5, 6, 7,
                                        8, 9, 10, 11, 12, 13, 14, 15);
      // B fragments (32x16): lane col, 16 contiguous K halves
      v16h b0 = *(const v16h*)(Bbuf + (bc + 0) * 64 + k_in + hi * 16);
      v16h b1 = *(const v16h*)(Bbuf + (bc + 16) * 64 + k_in + hi * 16);
      v16h b2 = *(const v16h*)(Bbuf + (bc + 32) * 64 + k_in + hi * 16);
      v16h b3 = *(const v16h*)(Bbuf + (bc + 48) * 64 + k_in + hi * 16);
      acc0 = __builtin_amdgcn_wmma_f32_16x16x32_f16(false, av, false, b0, (short)0, acc0, false, false);
      acc1 = __builtin_amdgcn_wmma_f32_16x16x32_f16(false, av, false, b1, (short)0, acc1, false, false);
      acc2 = __builtin_amdgcn_wmma_f32_16x16x32_f16(false, av, false, b2, (short)0, acc2, false, false);
      acc3 = __builtin_amdgcn_wmma_f32_16x16x32_f16(false, av, false, b3, (short)0, acc3, false, false);
    }
    __syncthreads();                          // done reading cur before overwrite
  }

  int m0 = m0b + (wid & 3) * 16;
  int n0 = n0b + (wid >> 2) * 64;
  v8f accs[4] = {acc0, acc1, acc2, acc3};
#pragma unroll
  for (int j = 0; j < 4; ++j) {
    int col = n0 + j * 16 + ln;
    float bv = bias ? bias[col] : 0.f;
#pragma unroll
    for (int r = 0; r < 8; ++r) {
      int row = m0 + r + hi * 8;  // C/D layout: lanes16-31 hold M=r+8
      float v = accs[j][r] + bv;
      if (do_gelu) v = gelu_exact(v);
      long idx = (long)row * N + col;
      if (resid) v += resid[idx];
      if (o32) o32[idx] = v;
      if (o16) o16[idx] = (f16)v;
    }
  }
}

// ------------------------------------------------------------------
// Self-attention per (b,h): 64 queries x 64 keys, dh=64, additive bias.
// qkv: [B*T, 3*D] f16 (q|k|v). out: [B*T, D] f16.
// ------------------------------------------------------------------
__global__ void __launch_bounds__(64)
k_attn(const f16* __restrict__ qkv, const float* __restrict__ logconf,
       f16* __restrict__ out) {
  __shared__ f16 Qs[TSEQ][DHEAD + 8];
  __shared__ f16 Ks[TSEQ][DHEAD + 8];
  __shared__ f16 Vs[TSEQ][DHEAD + 8];
  __shared__ float Bs[TSEQ];
  __shared__ float Ps[TSEQ][TSEQ + 1];
  int b = blockIdx.x / NHEAD;
  int h = blockIdx.x % NHEAD;
  int t = threadIdx.x;
  const f16* base = qkv + (long)(b * TSEQ + t) * (3 * DMODEL) + h * DHEAD;
  for (int d = 0; d < DHEAD; d += 8) {
    *(v8h*)&Qs[t][d] = *(const v8h*)(base + d);
    *(v8h*)&Ks[t][d] = *(const v8h*)(base + DMODEL + d);
    *(v8h*)&Vs[t][d] = *(const v8h*)(base + 2 * DMODEL + d);
  }
  Bs[t] = logconf[b * TSEQ + t];
  __syncthreads();
  float mx = -1e30f;
  for (int tk = 0; tk < TSEQ; ++tk) {
    float s = 0.f;
    for (int d = 0; d < DHEAD; ++d) s += (float)Qs[t][d] * (float)Ks[tk][d];
    s = s * 0.125f + Bs[tk];
    Ps[t][tk] = s;
    mx = fmaxf(mx, s);
  }
  float sum = 0.f;
  for (int tk = 0; tk < TSEQ; ++tk) {
    float e = __expf(Ps[t][tk] - mx);
    Ps[t][tk] = e;
    sum += e;
  }
  float inv = 1.f / sum;
  f16* orow = out + (long)(b * TSEQ + t) * DMODEL + h * DHEAD;
  for (int d = 0; d < DHEAD; ++d) {
    float a = 0.f;
    for (int tk = 0; tk < TSEQ; ++tk) a += Ps[t][tk] * (float)Vs[tk][d];
    orow[d] = (f16)(a * inv);
  }
}

// ------------------------------------------------------------------
// Pooled single-query attention per (b,h). qvec: [D] f32 (shared by all b).
// kv: [B*T, 3*D] f16 (k at +D, v at +2D). out: [B, D] f16. No bias.
// ------------------------------------------------------------------
__global__ void __launch_bounds__(64)
k_pooled_attn(const f16* __restrict__ kv, const float* __restrict__ qvec,
              f16* __restrict__ out) {
  __shared__ f16 Ks[TSEQ][DHEAD + 8];
  __shared__ f16 Vs[TSEQ][DHEAD + 8];
  __shared__ float sc[TSEQ];
  int b = blockIdx.x / NHEAD, h = blockIdx.x % NHEAD;
  int t = threadIdx.x;
  const f16* base = kv + (long)(b * TSEQ + t) * (3 * DMODEL) + h * DHEAD;
  for (int d = 0; d < DHEAD; d += 8) {
    *(v8h*)&Ks[t][d] = *(const v8h*)(base + DMODEL + d);
    *(v8h*)&Vs[t][d] = *(const v8h*)(base + 2 * DMODEL + d);
  }
  __syncthreads();
  float s = 0.f;
  for (int d = 0; d < DHEAD; ++d) s += qvec[h * DHEAD + d] * (float)Ks[t][d];
  sc[t] = s * 0.125f;
  __syncthreads();
  float mx = -1e30f;
  for (int tk = 0; tk < TSEQ; ++tk) mx = fmaxf(mx, sc[tk]);
  float sum = 0.f;
  for (int tk = 0; tk < TSEQ; ++tk) sum += __expf(sc[tk] - mx);
  float inv = 1.f / sum;
  int d = t;
  float a = 0.f;
  for (int tk = 0; tk < TSEQ; ++tk) a += __expf(sc[tk] - mx) * (float)Vs[tk][d];
  out[(long)b * DMODEL + h * DHEAD + d] = (f16)(a * inv);
}

// qvec[o] = query . wq[o,:] + bq[o]   (uses rows 0..D-1 of packed w_qkv)
__global__ void k_pooled_q(const float* __restrict__ query, const float* __restrict__ wqkv,
                           const float* __restrict__ bqkv, float* __restrict__ qvec) {
  int o = blockIdx.x * blockDim.x + threadIdx.x;
  if (o >= DMODEL) return;
  float s = bqkv[o];
  const float* wrow = wqkv + (long)o * DMODEL;
  for (int k = 0; k < DMODEL; ++k) s += query[k] * wrow[k];
  qvec[o] = s;
}

// dev[b,t,d] = x[b,t,d] - weighted_mean_t(x; conf)
__global__ void __launch_bounds__(256)
k_mean_dev(const float* __restrict__ x, const float* __restrict__ conf,
           f16* __restrict__ dev) {
  int b = blockIdx.x;
  float cs = 0.f;
  for (int t = 0; t < TSEQ; ++t) cs += conf[b * TSEQ + t];
  cs = fmaxf(cs, 1e-6f);
  for (int d = threadIdx.x; d < DMODEL; d += 256) {
    float s = 0.f;
    for (int t = 0; t < TSEQ; ++t)
      s += x[((long)b * TSEQ + t) * DMODEL + d] * conf[b * TSEQ + t];
    float mean = s / cs;
    for (int t = 0; t < TSEQ; ++t)
      dev[((long)b * TSEQ + t) * DMODEL + d] =
          (f16)(x[((long)b * TSEQ + t) * DMODEL + d] - mean);
  }
}

// concat(persistent, transient) -> f16 [B, 2D]
__global__ void k_concat(const float* __restrict__ p, const float* __restrict__ t,
                         f16* __restrict__ o) {
  int i = blockIdx.x * blockDim.x + threadIdx.x;
  if (i >= BBATCH * 2 * DMODEL) return;
  int r = i / (2 * DMODEL), c = i % (2 * DMODEL);
  float v = (c < DMODEL) ? p[(long)r * DMODEL + c] : t[(long)r * DMODEL + c - DMODEL];
  o[i] = (f16)v;
}

// ------------------------------------------------------------------
extern "C" void kernel_launch(void* const* d_in, const int* in_sizes, int n_in,
                              void* d_out, int out_size, void* d_ws, size_t ws_size,
                              hipStream_t stream) {
  (void)in_sizes; (void)n_in; (void)out_size; (void)ws_size;
  const float* frame   = (const float*)d_in[0];
  const float* ts      = (const float*)d_in[1];
  const float* cloud   = (const float*)d_in[2];
  const float* ln1_g   = (const float*)d_in[3];
  const float* ln1_b   = (const float*)d_in[4];
  const float* w_qkv   = (const float*)d_in[5];
  const float* b_qkv   = (const float*)d_in[6];
  const float* w_o     = (const float*)d_in[7];
  const float* b_o     = (const float*)d_in[8];
  const float* ln2_g   = (const float*)d_in[9];
  const float* ln2_b   = (const float*)d_in[10];
  const float* w1      = (const float*)d_in[11];
  const float* b1      = (const float*)d_in[12];
  const float* w2      = (const float*)d_in[13];
  const float* b2      = (const float*)d_in[14];
  const float* p_query = (const float*)d_in[15];
  const float* t_query = (const float*)d_in[16];
  const float* pw_qkv  = (const float*)d_in[17];
  const float* pb_qkv  = (const float*)d_in[18];
  const float* pw_o    = (const float*)d_in[19];
  const float* pb_o    = (const float*)d_in[20];
  const float* tw_qkv  = (const float*)d_in[21];
  const float* tb_qkv  = (const float*)d_in[22];
  const float* tw_o    = (const float*)d_in[23];
  const float* tb_o    = (const float*)d_in[24];
  const float* pn_g    = (const float*)d_in[25];
  const float* pn_b    = (const float*)d_in[26];
  const float* tn_g    = (const float*)d_in[27];
  const float* tn_b    = (const float*)d_in[28];
  const float* fw      = (const float*)d_in[29];
  const float* fb      = (const float*)d_in[30];
  const float* fln_g   = (const float*)d_in[31];
  const float* fln_b   = (const float*)d_in[32];

  float* out = (float*)d_out;
  float* out_fused = out;                             // [256,768]
  float* out_pers  = out + (long)BBATCH * DMODEL;     // [256,768]
  float* out_trans = out + 2L * BBATCH * DMODEL;      // [256,768]
  float* out_conf  = out + 3L * BBATCH * DMODEL;      // [256,64]

  // ---- workspace layout (bytes, 256B aligned) ----
  char* base = (char*)d_ws;
  size_t off = 0;
  auto alloc = [&](size_t bytes) -> size_t {
    size_t r = off;
    off += (bytes + 255) & ~(size_t)255;
    return r;
  };
  size_t o_wqkv16  = alloc((size_t)NLAYER * 3 * DMODEL * DMODEL * 2);
  size_t o_wo16    = alloc((size_t)NLAYER * DMODEL * DMODEL * 2);
  size_t o_w116    = alloc((size_t)NLAYER * DFF * DMODEL * 2);
  size_t o_w216    = alloc((size_t)NLAYER * DMODEL * DFF * 2);
  size_t o_pwqkv16 = alloc((size_t)3 * DMODEL * DMODEL * 2);
  size_t o_pwo16   = alloc((size_t)DMODEL * DMODEL * 2);
  size_t o_twqkv16 = alloc((size_t)3 * DMODEL * DMODEL * 2);
  size_t o_two16   = alloc((size_t)DMODEL * DMODEL * 2);
  size_t o_fw16    = alloc((size_t)DMODEL * 2 * DMODEL * 2);
  size_t o_x       = alloc((size_t)BT * DMODEL * 4);
  size_t o_ln16    = alloc((size_t)BT * DMODEL * 2);
  size_t o_act16   = alloc((size_t)BT * DFF * 2);
  size_t o_o16     = alloc((size_t)BT * DMODEL * 2);
  size_t o_conf    = alloc((size_t)BT * 4);
  size_t o_lconf   = alloc((size_t)BT * 4);
  size_t o_tmp     = alloc((size_t)BBATCH * DMODEL * 4);
  size_t o_qvec    = alloc((size_t)2 * DMODEL * 4);

  f16* wqkv16  = (f16*)(base + o_wqkv16);
  f16* wo16    = (f16*)(base + o_wo16);
  f16* w116    = (f16*)(base + o_w116);
  f16* w216    = (f16*)(base + o_w216);
  f16* pwqkv16 = (f16*)(base + o_pwqkv16);
  f16* pwo16   = (f16*)(base + o_pwo16);
  f16* twqkv16 = (f16*)(base + o_twqkv16);
  f16* two16   = (f16*)(base + o_two16);
  f16* fw16    = (f16*)(base + o_fw16);
  float* x     = (float*)(base + o_x);
  f16* ln16    = (f16*)(base + o_ln16);
  f16* act16   = (f16*)(base + o_act16);
  f16* o16     = (f16*)(base + o_o16);
  float* conf  = (float*)(base + o_conf);
  float* lconf = (float*)(base + o_lconf);
  float* tmp   = (float*)(base + o_tmp);
  float* qvecP = (float*)(base + o_qvec);
  float* qvecT = qvecP + DMODEL;

  const int CVT_GRID = 2048;
  // ---- convert weights to f16 ----
  k_cvt16<<<CVT_GRID, 256, 0, stream>>>(w_qkv, wqkv16, (long)NLAYER * 3 * DMODEL * DMODEL);
  k_cvt16<<<CVT_GRID, 256, 0, stream>>>(w_o, wo16, (long)NLAYER * DMODEL * DMODEL);
  k_cvt16<<<CVT_GRID, 256, 0, stream>>>(w1, w116, (long)NLAYER * DFF * DMODEL);
  k_cvt16<<<CVT_GRID, 256, 0, stream>>>(w2, w216, (long)NLAYER * DMODEL * DFF);
  k_cvt16<<<CVT_GRID, 256, 0, stream>>>(pw_qkv, pwqkv16, (long)3 * DMODEL * DMODEL);
  k_cvt16<<<CVT_GRID, 256, 0, stream>>>(pw_o, pwo16, (long)DMODEL * DMODEL);
  k_cvt16<<<CVT_GRID, 256, 0, stream>>>(tw_qkv, twqkv16, (long)3 * DMODEL * DMODEL);
  k_cvt16<<<CVT_GRID, 256, 0, stream>>>(tw_o, two16, (long)DMODEL * DMODEL);
  k_cvt16<<<CVT_GRID, 256, 0, stream>>>(fw, fw16, (long)DMODEL * 2 * DMODEL);

  // ---- embedding + confidence ----
  {
    long n = (long)BT * DMODEL;
    k_embed<<<(unsigned)((n + 255) / 256), 256, 0, stream>>>(frame, ts, x);
    k_conf<<<(BT + 255) / 256, 256, 0, stream>>>(cloud, conf, lconf, out_conf);
  }

  // ---- transformer layers ----
  for (int i = 0; i < NLAYER; ++i) {
    const f16* lw_qkv = wqkv16 + (long)i * 3 * DMODEL * DMODEL;
    const f16* lw_o   = wo16 + (long)i * DMODEL * DMODEL;
    const f16* lw_1   = w116 + (long)i * DFF * DMODEL;
    const f16* lw_2   = w216 + (long)i * DMODEL * DFF;

    // xn = LN1(x) -> f16
    k_layernorm<<<BT, 256, 0, stream>>>(x, ln1_g + i * DMODEL, ln1_b + i * DMODEL,
                                        (float*)nullptr, ln16);
    // qkv = xn @ w_qkv^T + b_qkv -> f16 [BT, 2304]
    k_gemm<<<dim3(3 * DMODEL / 128, BT / 64), 256, 0, stream>>>(
        ln16, lw_qkv, b_qkv + (long)i * 3 * DMODEL, (const float*)nullptr,
        (float*)nullptr, act16, BT, 3 * DMODEL, DMODEL, 0);
    // attention with additive log-confidence bias
    k_attn<<<BBATCH * NHEAD, 64, 0, stream>>>(act16, lconf, o16);
    // x = x + o @ w_o^T + b_o
    k_gemm<<<dim3(DMODEL / 128, BT / 64), 256, 0, stream>>>(
        o16, lw_o, b_o + (long)i * DMODEL, x, x, (f16*)nullptr,
        BT, DMODEL, DMODEL, 0);
    // hn = LN2(x) -> f16
    k_layernorm<<<BT, 256, 0, stream>>>(x, ln2_g + i * DMODEL, ln2_b + i * DMODEL,
                                        (float*)nullptr, ln16);
    // h1 = gelu(hn @ w1^T + b1) -> f16 [BT, 3072]
    k_gemm<<<dim3(DFF / 128, BT / 64), 256, 0, stream>>>(
        ln16, lw_1, b1 + (long)i * DFF, (const float*)nullptr,
        (float*)nullptr, act16, BT, DFF, DMODEL, 1);
    // x = x + h1 @ w2^T + b2
    k_gemm<<<dim3(DMODEL / 128, BT / 64), 256, 0, stream>>>(
        act16, lw_2, b2 + (long)i * DMODEL, x, x, (f16*)nullptr,
        BT, DMODEL, DFF, 0);
  }

  // ---- persistent pooled attention ----
  // x -> f16 (k/v source)
  k_cvt16<<<CVT_GRID, 256, 0, stream>>>(x, ln16, (long)BT * DMODEL);
  // kv = x @ pw_qkv^T + pb_qkv  (q portion unused)
  k_gemm<<<dim3(3 * DMODEL / 128, BT / 64), 256, 0, stream>>>(
      ln16, pwqkv16, pb_qkv, (const float*)nullptr, (float*)nullptr, act16,
      BT, 3 * DMODEL, DMODEL, 0);
  // q = p_query @ wq^T + bq
  k_pooled_q<<<3, 256, 0, stream>>>(p_query, pw_qkv, pb_qkv, qvecP);
  k_pooled_attn<<<BBATCH * NHEAD, 64, 0, stream>>>(act16, qvecP, o16);
  // out-proj -> tmp f32 [256, 768]
  k_gemm<<<dim3(DMODEL / 128, BBATCH / 64), 256, 0, stream>>>(
      o16, pwo16, pb_o, (const float*)nullptr, tmp, (f16*)nullptr,
      BBATCH, DMODEL, DMODEL, 0);
  // persistent = LN(tmp)
  k_layernorm<<<BBATCH, 256, 0, stream>>>(tmp, pn_g, pn_b, out_pers, (f16*)nullptr);

  // ---- transient pooled attention over deviations ----
  k_mean_dev<<<BBATCH, 256, 0, stream>>>(x, conf, ln16);
  k_gemm<<<dim3(3 * DMODEL / 128, BT / 64), 256, 0, stream>>>(
      ln16, twqkv16, tb_qkv, (const float*)nullptr, (float*)nullptr, act16,
      BT, 3 * DMODEL, DMODEL, 0);
  k_pooled_q<<<3, 256, 0, stream>>>(t_query, tw_qkv, tb_qkv, qvecT);
  k_pooled_attn<<<BBATCH * NHEAD, 64, 0, stream>>>(act16, qvecT, o16);
  k_gemm<<<dim3(DMODEL / 128, BBATCH / 64), 256, 0, stream>>>(
      o16, two16, tb_o, (const float*)nullptr, tmp, (f16*)nullptr,
      BBATCH, DMODEL, DMODEL, 0);
  k_layernorm<<<BBATCH, 256, 0, stream>>>(tmp, tn_g, tn_b, out_trans, (f16*)nullptr);

  // ---- fusion head ----
  k_concat<<<(BBATCH * 2 * DMODEL + 255) / 256, 256, 0, stream>>>(out_pers, out_trans, o16);
  // gelu(cat @ fw^T + fb) -> tmp
  k_gemm<<<dim3(DMODEL / 128, BBATCH / 64), 256, 0, stream>>>(
      o16, fw16, fb, (const float*)nullptr, tmp, (f16*)nullptr,
      BBATCH, DMODEL, 2 * DMODEL, 1);
  // fused = LN(tmp)
  k_layernorm<<<BBATCH, 256, 0, stream>>>(tmp, fln_g, fln_b, out_fused, (f16*)nullptr);
}
